// SAModule_77309412227
// MI455X (gfx1250) — matrix operand order repaired and checked by reading, not compiled, not run on hardware
//
#include <hip/hip_runtime.h>

// ---------------- problem constants ----------------
#define BB    2
#define NN    8192
#define CC    64
#define MM    2048
#define KNB   64
#define HID   97
#define HIDP  112            // 97 padded to 7*16
#define NOUT  128
#define NE    (BB*MM*KNB)    // 262144 edges
#define NRT   (NE/16)        // 16384 row tiles
#define R2C   0.01f
#define EPSBN 1e-5f
#define CAP   512
#define NCHUNK 64            // stats reduction chunks (NRT/NCHUNK = 256)

typedef __attribute__((ext_vector_type(16))) _Float16 v16h;
typedef __attribute__((ext_vector_type(8)))  _Float16 v8h;
typedef __attribute__((ext_vector_type(8)))  float    v8f;
typedef __attribute__((ext_vector_type(4)))  float    v4f;
union V16H { v16h v; v8h h8[2]; _Float16 h[16]; };
union V8F  { v8f  v; float    f[8];  };

// convert 8 contiguous f32 -> 8 f16 (two b128 loads, packed cvts)
__device__ __forceinline__ void ld8f(const float* __restrict__ p, _Float16* dst) {
  v4f u = *(const v4f*)p;
  v4f w = *(const v4f*)(p + 4);
#pragma unroll
  for (int q = 0; q < 4; ++q) { dst[q] = (_Float16)u[q]; dst[4 + q] = (_Float16)w[q]; }
}

// ---------------- 1) weight pre-swizzle into WMMA B layout (+ zero edge count) ----------------
// B operand layout (f16, KxN=32x16): lane L: col = L&15, hi = L>>4; half j -> K = (j<8 ? hi*8 + j : 16 + hi*8 + j-8).
__global__ void prep_w_kernel(const float* __restrict__ W1, const float* __restrict__ W2,
                              _Float16* __restrict__ w1swz, _Float16* __restrict__ w2swz,
                              int* __restrict__ cnti) {
  int tid = blockIdx.x * blockDim.x + threadIdx.x;
  int stride = gridDim.x * blockDim.x;
  if (tid == 0) *cnti = 0;
  for (int i = tid; i < 7 * 3 * 32 * 16; i += stride) {      // W1 (67x97)
    int j = i & 15; int t = i >> 4; int lane = t & 31; int t2 = t >> 5;
    int ks = t2 % 3, ct = t2 / 3;
    int hi = lane >> 4, c = lane & 15;
    int k = ks * 32 + (j < 8 ? hi * 8 + j : 16 + hi * 8 + (j - 8));
    int col = ct * 16 + c;
    float v = (k < CC + 3 && col < HID) ? W1[k * HID + col] : 0.f;
    w1swz[i] = (_Float16)v;
  }
  for (int i = tid; i < 8 * 4 * 32 * 16; i += stride) {      // W2 (97x128)
    int j = i & 15; int t = i >> 4; int lane = t & 31; int t2 = t >> 5;
    int ks = t2 & 3, ct = t2 >> 2;
    int hi = lane >> 4, c = lane & 15;
    int k = ks * 32 + (j < 8 ? hi * 8 + j : 16 + hi * 8 + (j - 8));
    int col = ct * 16 + c;
    float v = (k < HID) ? W2[k * NOUT + col] : 0.f;
    w2swz[i] = (_Float16)v;
  }
}

// ---------------- 2) farthest point sampling: one block per cloud, 2 barriers per step ----------------
__global__ __launch_bounds__(1024) void fps_kernel(const float* __restrict__ pos,
                                                   int* __restrict__ idx_ws, float* __restrict__ pos_sw,
                                                   int* __restrict__ idx_out, float* __restrict__ pos_out) {
  __shared__ float redv[32];
  __shared__ int   redi[32];
  __shared__ float cur[3];
  int b = blockIdx.x;
  int tid = threadIdx.x;
  int lane = tid & 31, wv = tid >> 5;
  const float* pb = pos + (size_t)b * NN * 3;
  float px[8], py[8], pz[8], md[8];
#pragma unroll
  for (int t = 0; t < 8; ++t) {
    int i = tid + t * 1024;
    px[t] = pb[i * 3 + 0]; py[t] = pb[i * 3 + 1]; pz[t] = pb[i * 3 + 2];
  }
  float x0 = pb[0], y0 = pb[1], z0 = pb[2];
#pragma unroll
  for (int t = 0; t < 8; ++t) {
    float dx = px[t] - x0, dy = py[t] - y0, dz = pz[t] - z0;
    md[t] = dx * dx + dy * dy + dz * dz;
  }
  if (tid == 0) {
    idx_ws[b * MM] = 0; idx_out[b * MM] = 0;
    size_t o = (size_t)(b * MM) * 3;
    pos_sw[o] = x0; pos_sw[o + 1] = y0; pos_sw[o + 2] = z0;
    pos_out[o] = x0; pos_out[o + 1] = y0; pos_out[o + 2] = z0;
  }
  __syncthreads();
  for (int s = 1; s < MM; ++s) {
    // local argmax over this thread's 8 strided points (lowest index wins ties)
    float bv = -1.f; int bi = NN;
#pragma unroll
    for (int t = 0; t < 8; ++t) {
      int i = tid + t * 1024;
      if (md[t] > bv || (md[t] == bv && i < bi)) { bv = md[t]; bi = i; }
    }
    // wave butterfly (no barrier)
#pragma unroll
    for (int off = 16; off > 0; off >>= 1) {
      float ov = __shfl_xor(bv, off, 32);
      int   oi = __shfl_xor(bi, off, 32);
      if (ov > bv || (ov == bv && oi < bi)) { bv = ov; bi = oi; }
    }
    if (lane == 0) { redv[wv] = bv; redi[wv] = bi; }
    __syncthreads();
    if (wv == 0) {
      float v2 = redv[lane]; int i2 = redi[lane];
#pragma unroll
      for (int off = 16; off > 0; off >>= 1) {
        float ov = __shfl_xor(v2, off, 32);
        int   oi = __shfl_xor(i2, off, 32);
        if (ov > v2 || (ov == v2 && oi < i2)) { v2 = ov; i2 = oi; }
      }
      if (lane == 0) {
        int j = i2;
        idx_ws[b * MM + s] = j; idx_out[b * MM + s] = j;
        float cx = pb[j * 3], cy = pb[j * 3 + 1], cz = pb[j * 3 + 2];
        cur[0] = cx; cur[1] = cy; cur[2] = cz;
        size_t o = (size_t)(b * MM + s) * 3;
        pos_sw[o] = cx; pos_sw[o + 1] = cy; pos_sw[o + 2] = cz;
        pos_out[o] = cx; pos_out[o + 1] = cy; pos_out[o + 2] = cz;
      }
    }
    __syncthreads();
    float cx = cur[0], cy = cur[1], cz = cur[2];
#pragma unroll
    for (int t = 0; t < 8; ++t) {
      float dx = px[t] - cx, dy = py[t] - cy, dz = pz[t] - cz;
      float d = dx * dx + dy * dy + dz * dz;
      if (d < md[t]) md[t] = d;
    }
  }
}

// ---------------- 3) radius ball query, K nearest within R: one wave per centroid ----------------
__global__ __launch_bounds__(256) void ballquery_kernel(const float* __restrict__ pos,
                                                        const float* __restrict__ pos_sw,
                                                        int* __restrict__ nidx, float* __restrict__ validf,
                                                        int* __restrict__ cnti) {
  __shared__ float cd2[8][CAP];
  __shared__ int   cix[8][CAP];
  __shared__ int   scnt[8];
  int w = threadIdx.x >> 5, lane = threadIdx.x & 31;
  int cid = blockIdx.x * 8 + w;
  int b = cid / MM;
  if (lane == 0) scnt[w] = 0;
  __syncthreads();
  float cx = pos_sw[(size_t)cid * 3], cy = pos_sw[(size_t)cid * 3 + 1], cz = pos_sw[(size_t)cid * 3 + 2];
  const float* pb = pos + (size_t)b * NN * 3;
  for (int i = lane; i < NN; i += 32) {
    float dx = pb[i * 3] - cx, dy = pb[i * 3 + 1] - cy, dz = pb[i * 3 + 2] - cz;
    float d2 = dx * dx + dy * dy + dz * dz;
    if (d2 <= R2C) {
      int slot = atomicAdd(&scnt[w], 1);
      if (slot < CAP) { cd2[w][slot] = d2; cix[w][slot] = i; }
    }
  }
  __syncthreads();
  int n = scnt[w]; if (n > CAP) n = CAP;
  int*   no = nidx   + (size_t)cid * KNB;
  float* vo = validf + (size_t)cid * KNB;
  if (n <= KNB) {
    for (int k = lane; k < KNB; k += 32) {
      if (k < n) { no[k] = cix[w][k]; vo[k] = 1.f; }
      else       { no[k] = 0;         vo[k] = 0.f; }
    }
    if (lane == 0) atomicAdd(cnti, n);
  } else {
    for (int k = 0; k < KNB; ++k) {   // extract K smallest (d2, idx)
      float bd = 3.0e38f; int bi = 1 << 30; int bs = 0;
      for (int s = lane; s < n; s += 32) {
        float d = cd2[w][s]; int ii = cix[w][s];
        if (d < bd || (d == bd && ii < bi)) { bd = d; bi = ii; bs = s; }
      }
#pragma unroll
      for (int off = 16; off > 0; off >>= 1) {
        float od = __shfl_xor(bd, off, 32);
        int   oi = __shfl_xor(bi, off, 32);
        int   os = __shfl_xor(bs, off, 32);
        if (od < bd || (od == bd && oi < bi)) { bd = od; bi = oi; bs = os; }
      }
      if (lane == 0) { cd2[w][bs] = 3.0e38f; no[k] = bi; vo[k] = 1.f; }
    }
    if (lane == 0) atomicAdd(cnti, KNB);
  }
}

// ---------------- 4) GEMM1: gather(x||rel) @ W1 -> relu -> H1 (f16), masked partial stats ----------------
// One wave per 16-edge row tile; A gathered once, ct fully unrolled -> all addresses are base+imm.
__global__ __launch_bounds__(256) void gemm1_kernel(const float* __restrict__ x, const float* __restrict__ pos,
                                                    const float* __restrict__ pos_sw,
                                                    const int* __restrict__ nidx, const float* __restrict__ validf,
                                                    const _Float16* __restrict__ w1swz, const float* __restrict__ b1,
                                                    _Float16* __restrict__ H1,
                                                    float* __restrict__ p1s, float* __restrict__ p1q) {
  int rt   = ((blockIdx.x * 256) + threadIdx.x) >> 5;
  int lane = threadIdx.x & 31;
  int hi = lane >> 4, r = lane & 15;
  int e0 = rt * 16;
  int e  = e0 + r;
  int b  = e >> 17;           // / (M*K)
  int m  = (e & 131071) >> 6;
  int j  = nidx[e];           // invalid rows use j=0: finite garbage, masked later
  float c0 = pos_sw[(size_t)(b * MM + m) * 3 + 0];
  float c1 = pos_sw[(size_t)(b * MM + m) * 3 + 1];
  float c2 = pos_sw[(size_t)(b * MM + m) * 3 + 2];
  const float* xr = x   + ((size_t)b * NN + j) * CC;
  const float* pr = pos + ((size_t)b * NN + j) * 3;

  // A operand: halves [0..7] -> f = ks*32 + hi*8 + q ; [8..15] -> f = ks*32 + 16 + hi*8 + q
  V16H A0, A1, A2;
  ld8f(xr +      hi * 8, &A0.h[0]);
  ld8f(xr + 16 + hi * 8, &A0.h[8]);
  ld8f(xr + 32 + hi * 8, &A1.h[0]);
  ld8f(xr + 48 + hi * 8, &A1.h[8]);
#pragma unroll
  for (int q = 0; q < 16; ++q) A2.h[q] = (_Float16)0.f;
  if (hi == 0) {               // f = 64,65,66 -> rel; rest of k-step 2 is zero padding
    A2.h[0] = (_Float16)(pr[0] - c0);
    A2.h[1] = (_Float16)(pr[1] - c1);
    A2.h[2] = (_Float16)(pr[2] - c2);
  }
  // validity mask for the 8 rows this lane owns in C-layout (rows hi*8 .. hi*8+7)
  v4f mk0 = *(const v4f*)(validf + e0 + 8 * hi);
  v4f mk1 = *(const v4f*)(validf + e0 + 8 * hi + 4);
  float mk[8];
#pragma unroll
  for (int q = 0; q < 4; ++q) { mk[q] = mk0[q]; mk[4 + q] = mk1[q]; }

  // preload biases (7 clause-able loads, imm offsets off b1+r)
  const float* bb = b1 + r;
  float bias[7];
#pragma unroll
  for (int ct = 0; ct < 7; ++ct) bias[ct] = (ct * 16 + r < HID) ? bb[ct * 16] : 0.f;

  // base pointers: every access below is base + compile-time-constant offset
  const _Float16* wb   = w1swz + (size_t)lane * 16;                       // + (ct*3+ks)*512
  _Float16*       hstb = H1 + (size_t)(e0 + 8 * hi) * HIDP + r;           // + v*HIDP + ct*16
  float*          psb  = p1s + (size_t)rt * HIDP + r;                     // + ct*16
  float*          pqb  = p1q + (size_t)rt * HIDP + r;

#pragma unroll
  for (int ct = 0; ct < 7; ++ct) {
    V8F acc;
#pragma unroll
    for (int i = 0; i < 8; ++i) acc.f[i] = 0.f;
    V16H bm;
    bm.v = *(const v16h*)(wb + (ct * 3 + 0) * 512);
    acc.v = __builtin_amdgcn_wmma_f32_16x16x32_f16(false, A0.v, false, bm.v, (short)0, acc.v, false, false);
    bm.v = *(const v16h*)(wb + (ct * 3 + 1) * 512);
    acc.v = __builtin_amdgcn_wmma_f32_16x16x32_f16(false, A1.v, false, bm.v, (short)0, acc.v, false, false);
    bm.v = *(const v16h*)(wb + (ct * 3 + 2) * 512);
    acc.v = __builtin_amdgcn_wmma_f32_16x16x32_f16(false, A2.v, false, bm.v, (short)0, acc.v, false, false);

    float s = 0.f, q2 = 0.f;
#pragma unroll
    for (int v = 0; v < 8; ++v) {
      float h = acc.f[v] + bias[ct]; h = h > 0.f ? h : 0.f;
      hstb[v * HIDP + ct * 16] = (_Float16)h;
      s += h * mk[v]; q2 += h * h * mk[v];
    }
    s  += __shfl_xor(s, 16, 32);
    q2 += __shfl_xor(q2, 16, 32);
    if (lane < 16) { psb[ct * 16] = s; pqb[ct * 16] = q2; }
  }
}

// ---------------- 5a) deterministic partial reduction of per-tile stats ----------------
__global__ void reduce_part_kernel(const float* __restrict__ ps, const float* __restrict__ pq,
                                   float* __restrict__ ops, float* __restrict__ opq, int pitch) {
  int c = threadIdx.x;
  if (c >= pitch) return;
  int r0 = blockIdx.x * (NRT / NCHUNK);
  float s = 0.f, q = 0.f;
  for (int t = 0; t < NRT / NCHUNK; ++t) {
    s += ps[(size_t)(r0 + t) * pitch + c];
    q += pq[(size_t)(r0 + t) * pitch + c];
  }
  ops[blockIdx.x * pitch + c] = s;
  opq[blockIdx.x * pitch + c] = q;
}

// ---------------- 5b) finalize BN affine: scale = g*rsqrt(var+eps), shift = be - mean*scale ----------------
__global__ void finalize_kernel(const float* __restrict__ pps, const float* __restrict__ ppq,
                                const int* __restrict__ cnti,
                                const float* __restrict__ g, const float* __restrict__ be,
                                float* __restrict__ scale, float* __restrict__ shift,
                                int pitch, int nreal) {
  int c = threadIdx.x;
  if (c >= pitch) return;
  float s = 0.f, q = 0.f;
  for (int t = 0; t < NCHUNK; ++t) { s += pps[t * pitch + c]; q += ppq[t * pitch + c]; }
  if (c < nreal) {
    float n = (float)cnti[0]; if (n < 1.f) n = 1.f;
    float mu = s / n;
    float var = q / n - mu * mu;
    float sc = g[c] * rsqrtf(var + EPSBN);
    scale[c] = sc; shift[c] = be[c] - mu * sc;
  } else { scale[c] = 0.f; shift[c] = 0.f; }
}

// ---------------- 5c) apply BN1 to H1 in place (hoists affine out of GEMM2/final A-loads) ----------------
__global__ __launch_bounds__(256) void bn1apply_kernel(_Float16* __restrict__ H1,
                                                       const float* __restrict__ sc1,
                                                       const float* __restrict__ sh1) {
  size_t i = ((size_t)blockIdx.x * 256 + threadIdx.x) * 8;   // 8 halves / thread; HIDP % 8 == 0
  int col0 = (int)(i % HIDP);
  V16H t; t.h8[0] = *(const v8h*)(H1 + i);
  v4f s0 = *(const v4f*)(sc1 + col0),  s1 = *(const v4f*)(sc1 + col0 + 4);
  v4f h0 = *(const v4f*)(sh1 + col0),  h1 = *(const v4f*)(sh1 + col0 + 4);
  V16H u;
#pragma unroll
  for (int q = 0; q < 4; ++q) {
    u.h[q]     = (_Float16)((float)t.h[q]     * s0[q] + h0[q]);
    u.h[4 + q] = (_Float16)((float)t.h[4 + q] * s1[q] + h1[q]);
  }
  *(v8h*)(H1 + i) = u.h8[0];
}

// ---------------- 6) GEMM2 stats: relu(H1n @ W2 + b2), masked partial stats (A reused over 8 col tiles) ----
__global__ __launch_bounds__(256) void gemm2_stats_kernel(const _Float16* __restrict__ H1,
                                                          const _Float16* __restrict__ w2swz,
                                                          const float* __restrict__ b2,
                                                          const float* __restrict__ validf,
                                                          float* __restrict__ p2s, float* __restrict__ p2q) {
  int rt   = ((blockIdx.x * 256) + threadIdx.x) >> 5;
  int lane = threadIdx.x & 31;
  int hi = lane >> 4, r = lane & 15;
  int e0 = rt * 16;
  const _Float16* hr = H1 + (size_t)(e0 + r) * HIDP + hi * 8;
  V16H A[4];
#pragma unroll
  for (int ks = 0; ks < 4; ++ks) {
    A[ks].h8[0] = *(const v8h*)(hr + ks * 32);
    if (ks < 3) A[ks].h8[1] = *(const v8h*)(hr + ks * 32 + 16);
    else {
#pragma unroll
      for (int q = 0; q < 8; ++q) A[3].h[8 + q] = (_Float16)0.f;   // K 112..127 pad
    }
  }
  v4f mk0 = *(const v4f*)(validf + e0 + 8 * hi);
  v4f mk1 = *(const v4f*)(validf + e0 + 8 * hi + 4);
  float mk[8];
#pragma unroll
  for (int q = 0; q < 4; ++q) { mk[q] = mk0[q]; mk[4 + q] = mk1[q]; }

  const float* bb = b2 + r;
  float bias[8];
#pragma unroll
  for (int ct = 0; ct < 8; ++ct) bias[ct] = bb[ct * 16];

  const _Float16* wb  = w2swz + (size_t)lane * 16;        // + (ct*4+ks)*512
  float*          psb = p2s + (size_t)rt * NOUT + r;      // + ct*16
  float*          pqb = p2q + (size_t)rt * NOUT + r;

#pragma unroll
  for (int ct = 0; ct < 8; ++ct) {
    V8F acc;
#pragma unroll
    for (int i = 0; i < 8; ++i) acc.f[i] = 0.f;
#pragma unroll
    for (int ks = 0; ks < 4; ++ks) {
      V16H bm; bm.v = *(const v16h*)(wb + (ct * 4 + ks) * 512);
      acc.v = __builtin_amdgcn_wmma_f32_16x16x32_f16(false, A[ks].v, false, bm.v, (short)0, acc.v, false, false);
    }
    float s = 0.f, q2 = 0.f;
#pragma unroll
    for (int v = 0; v < 8; ++v) {
      float h = acc.f[v] + bias[ct]; h = h > 0.f ? h : 0.f;
      s += h * mk[v]; q2 += h * h * mk[v];
    }
    s  += __shfl_xor(s, 16, 32);
    q2 += __shfl_xor(q2, 16, 32);
    if (lane < 16) { psb[ct * 16] = s; pqb[ct * 16] = q2; }
  }
}

// ---------------- 8) final: recompute GEMM2, apply BN2, masked max over K -> x_out ----------------
__global__ __launch_bounds__(256) void final_kernel(const _Float16* __restrict__ H1,
                                                    const _Float16* __restrict__ w2swz,
                                                    const float* __restrict__ b2,
                                                    const float* __restrict__ sc2, const float* __restrict__ sh2,
                                                    const float* __restrict__ validf, float* __restrict__ xout) {
  int cid  = ((blockIdx.x * 256) + threadIdx.x) >> 5;   // one wave per centroid
  int lane = threadIdx.x & 31;
  int hi = lane >> 4, r = lane & 15;
  float mx[8];
#pragma unroll
  for (int ct = 0; ct < 8; ++ct) mx[ct] = -3.0e38f;

  float bias[8], sc[8], sh[8];
#pragma unroll
  for (int ct = 0; ct < 8; ++ct) {
    bias[ct] = (b2 + r)[ct * 16];
    sc[ct]   = (sc2 + r)[ct * 16];
    sh[ct]   = (sh2 + r)[ct * 16];
  }
  const _Float16* wb = w2swz + (size_t)lane * 16;        // + (ct*4+ks)*512

  for (int rtl = 0; rtl < 4; ++rtl) {
    int e0 = cid * KNB + rtl * 16;
    const _Float16* hr = H1 + (size_t)(e0 + r) * HIDP + hi * 8;
    V16H A[4];
#pragma unroll
    for (int ks = 0; ks < 4; ++ks) {
      A[ks].h8[0] = *(const v8h*)(hr + ks * 32);
      if (ks < 3) A[ks].h8[1] = *(const v8h*)(hr + ks * 32 + 16);
      else {
#pragma unroll
        for (int q = 0; q < 8; ++q) A[3].h[8 + q] = (_Float16)0.f;
      }
    }
    v4f mk0 = *(const v4f*)(validf + e0 + 8 * hi);
    v4f mk1 = *(const v4f*)(validf + e0 + 8 * hi + 4);
    float mk[8];
#pragma unroll
    for (int q = 0; q < 4; ++q) { mk[q] = mk0[q]; mk[4 + q] = mk1[q]; }

#pragma unroll
    for (int ct = 0; ct < 8; ++ct) {
      V8F acc;
#pragma unroll
      for (int i = 0; i < 8; ++i) acc.f[i] = 0.f;
#pragma unroll
      for (int ks = 0; ks < 4; ++ks) {
        V16H bm; bm.v = *(const v16h*)(wb + (ct * 4 + ks) * 512);
        acc.v = __builtin_amdgcn_wmma_f32_16x16x32_f16(false, A[ks].v, false, bm.v, (short)0, acc.v, false, false);
      }
#pragma unroll
      for (int v = 0; v < 8; ++v) {
        if (mk[v] > 0.f) {
          float h = acc.f[v] + bias[ct]; h = h > 0.f ? h : 0.f;
          float hn = h * sc[ct] + sh[ct];
          if (hn > mx[ct]) mx[ct] = hn;
        }
      }
    }
  }
  float* xb = xout + (size_t)cid * NOUT + r;
#pragma unroll
  for (int ct = 0; ct < 8; ++ct) {
    float o = __shfl_xor(mx[ct], 16, 32);
    float v = mx[ct] > o ? mx[ct] : o;
    if (lane < 16) xb[ct * 16] = v;
  }
}

// ---------------- 9) batch_s gather ----------------
__global__ void batchs_kernel(const int* __restrict__ batch, const int* __restrict__ idxs,
                              int* __restrict__ outb) {
  int i = blockIdx.x * blockDim.x + threadIdx.x;
  if (i < BB * MM) {
    int b = i / MM;
    outb[i] = batch[b * NN + idxs[i]];
  }
}

// ---------------- host launch ----------------
extern "C" void kernel_launch(void* const* d_in, const int* in_sizes, int n_in,
                              void* d_out, int out_size, void* d_ws, size_t ws_size,
                              hipStream_t stream) {
  const float* x    = (const float*)d_in[0];
  const float* pos  = (const float*)d_in[1];
  const int*   batch= (const int*)  d_in[2];
  const float* W1   = (const float*)d_in[3];
  const float* b1   = (const float*)d_in[4];
  const float* g1   = (const float*)d_in[5];
  const float* be1  = (const float*)d_in[6];
  const float* W2   = (const float*)d_in[7];
  const float* b2   = (const float*)d_in[8];
  const float* g2   = (const float*)d_in[9];
  const float* be2  = (const float*)d_in[10];
  (void)in_sizes; (void)n_in; (void)out_size; (void)ws_size;

  char* wp = (char*)d_ws;
  auto carve = [&](size_t bytes) -> void* {
    void* p = (void*)wp;
    wp += (bytes + 255) & ~(size_t)255;
    return p;
  };
  int*       idx_s  = (int*)      carve((size_t)BB * MM * 4);
  float*     pos_sw = (float*)    carve((size_t)BB * MM * 3 * 4);
  int*       nidx   = (int*)      carve((size_t)NE * 4);
  float*     validf = (float*)    carve((size_t)NE * 4);
  _Float16*  w1swz  = (_Float16*) carve((size_t)7 * 3 * 32 * 16 * 2);
  _Float16*  w2swz  = (_Float16*) carve((size_t)8 * 4 * 32 * 16 * 2);
  int*       cnti   = (int*)      carve(256);
  float*     sc1    = (float*)    carve((size_t)HIDP * 4);
  float*     sh1    = (float*)    carve((size_t)HIDP * 4);
  float*     sc2    = (float*)    carve((size_t)NOUT * 4);
  float*     sh2    = (float*)    carve((size_t)NOUT * 4);
  float*     pps    = (float*)    carve((size_t)NCHUNK * NOUT * 4);   // reused for both BN passes
  float*     ppq    = (float*)    carve((size_t)NCHUNK * NOUT * 4);
  float*     p1s    = (float*)    carve((size_t)NRT * HIDP * 4);
  float*     p1q    = (float*)    carve((size_t)NRT * HIDP * 4);
  float*     p2s    = (float*)    carve((size_t)NRT * NOUT * 4);
  float*     p2q    = (float*)    carve((size_t)NRT * NOUT * 4);
  _Float16*  H1     = (_Float16*) carve((size_t)NE * HIDP * 2);

  // output layout: x_out | pos_s | batch_s | idx
  float* xout      = (float*)d_out;
  float* pos_out   = xout + (size_t)BB * MM * NOUT;
  int*   batch_out = (int*)(pos_out + (size_t)BB * MM * 3);
  int*   idx_out   = batch_out + BB * MM;

  prep_w_kernel<<<8, 256, 0, stream>>>(W1, W2, w1swz, w2swz, cnti);
  fps_kernel<<<BB, 1024, 0, stream>>>(pos, idx_s, pos_sw, idx_out, pos_out);
  ballquery_kernel<<<(BB * MM) / 8, 256, 0, stream>>>(pos, pos_sw, nidx, validf, cnti);
  gemm1_kernel<<<NRT / 8, 256, 0, stream>>>(x, pos, pos_sw, nidx, validf, w1swz, b1, H1, p1s, p1q);
  reduce_part_kernel<<<NCHUNK, HIDP, 0, stream>>>(p1s, p1q, pps, ppq, HIDP);
  finalize_kernel<<<1, HIDP, 0, stream>>>(pps, ppq, cnti, g1, be1, sc1, sh1, HIDP, HID);
  bn1apply_kernel<<<(int)(((size_t)NE * HIDP / 8) / 256), 256, 0, stream>>>(H1, sc1, sh1);
  gemm2_stats_kernel<<<NRT / 8, 256, 0, stream>>>(H1, w2swz, b2, validf, p2s, p2q);
  reduce_part_kernel<<<NCHUNK, NOUT, 0, stream>>>(p2s, p2q, pps, ppq, NOUT);
  finalize_kernel<<<1, NOUT, 0, stream>>>(pps, ppq, cnti, g2, be2, sc2, sh2, NOUT, NOUT);
  final_kernel<<<(BB * MM) / 8, 256, 0, stream>>>(H1, w2swz, b2, sc2, sh2, validf, xout);
  batchs_kernel<<<(BB * MM + 255) / 256, 256, 0, stream>>>(batch, idx_s, batch_out);
}
